// ClassCenters_22376779612174
// MI455X (gfx1250) — matrix operand order repaired and checked by reading, not compiled, not run on hardware
//
#include <hip/hip_runtime.h>
#include <hip/hip_bf16.h>

// ---------------------------------------------------------------------------
// Pairwise squared distances: out[N,C] = max(|x|^2 + |y|^2 - 2 x.y, 0)
// N=16384, C=1000, D=1024.
// Pass 1: split f32 -> bf16 hi/lo arrays (+ row norms) in workspace.
// Pass 2: bf16 WMMA GEMM, x.y ~= Ah*Bh + Ah*Bl + Al*Bh, f32 accumulate.
// Wave tile 64(M)x64(N): 48 WMMAs per 32-deep K-step vs 32 b128 loads
// (49 FLOP per L2 byte -> compute-bound against the WMMA pipe).
// ---------------------------------------------------------------------------

#define DIM_N 16384
#define DIM_C 1000
#define DIM_D 1024

typedef __attribute__((ext_vector_type(16))) __bf16 v16bf;
typedef __attribute__((ext_vector_type(8)))  float  v8f;

// Round-to-nearest-even f32 -> bf16 bits; residual returned via *res.
static __device__ inline unsigned short bf16h(float f, float* res) {
    unsigned u = __builtin_bit_cast(unsigned, f);
    unsigned r = u + 0x7FFFu + ((u >> 16) & 1u);
    unsigned short h = (unsigned short)(r >> 16);
    float fh = __builtin_bit_cast(float, (unsigned)h << 16);
    *res = f - fh;
    return h;
}

// A fragment: two 8-bf16 runs at p and p+16 elements (each 16B, aligned).
static __device__ inline v16bf ldfragA(const unsigned short* p) {
    union { v16bf v; uint4 q[2]; } u;
    u.q[0] = *(const uint4*)(p);
    u.q[1] = *(const uint4*)(p + 16);
    return u.v;
}
// B fragment: two 8-bf16 runs at p and p+16 elements (each 16B, aligned).
static __device__ inline v16bf ldfragB(const unsigned short* p) {
    union { v16bf v; uint4 q[2]; } u;
    u.q[0] = *(const uint4*)(p);
    u.q[1] = *(const uint4*)(p + 16);
    return u.v;
}

// ---------------------------------------------------------------------------
// Split + row-norm kernel: one wave per row of X [rows x 1024] f32.
// Writes Xh, Xl (bf16 bits, row-major) and norms[row] = sum(x*x).
// ---------------------------------------------------------------------------
__global__ void split_norm_kernel(const float* __restrict__ X,
                                  unsigned short* __restrict__ Xh,
                                  unsigned short* __restrict__ Xl,
                                  float* __restrict__ norms, int rows) {
    const int wave = threadIdx.x >> 5;
    const int lane = threadIdx.x & 31;
    const int row  = blockIdx.x * (blockDim.x >> 5) + wave;
    if (row >= rows) return;

    const float* p = X + (size_t)row * DIM_D;
    unsigned short* ph = Xh + (size_t)row * DIM_D;
    unsigned short* pl = Xl + (size_t)row * DIM_D;

    float s = 0.0f;
#pragma unroll
    for (int i = 0; i < DIM_D / 128; ++i) {            // 8 iters of float4
        const int off = lane * 4 + i * 128;
        float4 v = *(const float4*)(p + off);
        s += v.x * v.x + v.y * v.y + v.z * v.z + v.w * v.w;
        float r0, r1, r2, r3, dmy;
        unsigned h0 = bf16h(v.x, &r0), h1 = bf16h(v.y, &r1);
        unsigned h2 = bf16h(v.z, &r2), h3 = bf16h(v.w, &r3);
        unsigned l0 = bf16h(r0, &dmy), l1 = bf16h(r1, &dmy);
        unsigned l2 = bf16h(r2, &dmy), l3 = bf16h(r3, &dmy);
        uint2 hp, lp;
        hp.x = h0 | (h1 << 16); hp.y = h2 | (h3 << 16);
        lp.x = l0 | (l1 << 16); lp.y = l2 | (l3 << 16);
        *(uint2*)(ph + off) = hp;                      // 8B aligned
        *(uint2*)(pl + off) = lp;
    }
#pragma unroll
    for (int m = 16; m >= 1; m >>= 1)
        s += __shfl_xor(s, m, 32);
    if (lane == 0) norms[row] = s;
}

// ---------------------------------------------------------------------------
// GEMM + epilogue. Block = 128 threads = 4 waves; each wave owns 64(M)x64(N).
// All 4 waves share n0 -> identical B fragments hit in WGP$.
// Grid: (DIM_N/256, 1024/64) = (64, 16).
// ---------------------------------------------------------------------------
__global__ void __launch_bounds__(128)
dist_wmma_kernel(const unsigned short* __restrict__ Ah,
                 const unsigned short* __restrict__ Al,
                 const unsigned short* __restrict__ Bh,
                 const unsigned short* __restrict__ Bl,
                 const float* __restrict__ anorm,
                 const float* __restrict__ bnorm,
                 float* __restrict__ out) {
    const int wave  = threadIdx.x >> 5;
    const int lane  = threadIdx.x & 31;
    const int lmod  = lane & 15;
    const int lhalf = lane >> 4;           // 0 or 1
    const int khalfA = lhalf * 8;          // A-fragment K sub-offset
    const int kbaseB = lhalf * 8;          // B-fragment K sub-offset

    const int m0 = (blockIdx.x * 4 + wave) * 64;   // 64 rows per wave
    const int n0 = blockIdx.y * 64;                // 64 cols per wave

    // A fragment base pointers for the four 16-row sub-tiles.
    const unsigned short* paH[4];
    const unsigned short* paL[4];
#pragma unroll
    for (int mt = 0; mt < 4; ++mt) {
        const size_t rowoff = (size_t)(m0 + mt * 16 + lmod) * DIM_D + khalfA;
        paH[mt] = Ah + rowoff;
        paL[mt] = Al + rowoff;
    }
    // B fragment base pointers for 4 column sub-tiles (clamped ragged edge;
    // keeps EXEC all-ones through the WMMAs; stores masked in epilogue).
    const unsigned short* pbH[4];
    const unsigned short* pbL[4];
#pragma unroll
    for (int nt = 0; nt < 4; ++nt) {
        int col = n0 + nt * 16 + lmod;
        if (col >= DIM_C) col = DIM_C - 1;
        const size_t coloff = (size_t)col * DIM_D + kbaseB;
        pbH[nt] = Bh + coloff;
        pbL[nt] = Bl + coloff;
    }

    v8f acc[4][4];
#pragma unroll
    for (int mt = 0; mt < 4; ++mt)
#pragma unroll
        for (int nt = 0; nt < 4; ++nt)
            acc[mt][nt] = (v8f){0.f, 0.f, 0.f, 0.f, 0.f, 0.f, 0.f, 0.f};

    for (int kt = 0; kt < DIM_D; kt += 32) {
        v16bf ah[4], al[4];
#pragma unroll
        for (int mt = 0; mt < 4; ++mt) {
            ah[mt] = ldfragA(paH[mt] + kt);
            al[mt] = ldfragA(paL[mt] + kt);
        }
#pragma unroll
        for (int nt = 0; nt < 4; ++nt) {
            const v16bf bh = ldfragB(pbH[nt] + kt);
            const v16bf bl = ldfragB(pbL[nt] + kt);
#pragma unroll
            for (int mt = 0; mt < 4; ++mt) {
                v8f c = acc[mt][nt];
                c = __builtin_amdgcn_wmma_f32_16x16x32_bf16(
                        false, al[mt], false, bh, (short)0, c, false, false);
                c = __builtin_amdgcn_wmma_f32_16x16x32_bf16(
                        false, ah[mt], false, bl, (short)0, c, false, false);
                c = __builtin_amdgcn_wmma_f32_16x16x32_bf16(
                        false, ah[mt], false, bh, (short)0, c, false, false);
                acc[mt][nt] = c;
            }
        }
    }

    // ---- Epilogue: dist = max(|x|^2 + |y|^2 - 2*dot, 0) ----
#pragma unroll
    for (int mt = 0; mt < 4; ++mt) {
#pragma unroll
        for (int r = 0; r < 8; ++r) {
            const int m = m0 + mt * 16 + r + lhalf * 8;   // C/D layout: VGPR r
            const float an = anorm[m];
#pragma unroll
            for (int nt = 0; nt < 4; ++nt) {
                const int n = n0 + nt * 16 + lmod;
                if (n < DIM_C) {
                    const float d = an + bnorm[n] - 2.0f * acc[mt][nt][r];
                    out[(size_t)m * DIM_C + n] = fmaxf(d, 0.0f);
                }
            }
        }
    }
}

// ---------------------------------------------------------------------------
extern "C" void kernel_launch(void* const* d_in, const int* in_sizes, int n_in,
                              void* d_out, int out_size, void* d_ws, size_t ws_size,
                              hipStream_t stream) {
    (void)in_sizes; (void)n_in; (void)out_size; (void)ws_size;
    const float* emb = (const float*)d_in[0];   // [16384, 1024] f32
    const float* cen = (const float*)d_in[1];   // [1000, 1024]  f32
    float* out = (float*)d_out;                 // [16384, 1000] f32

    // Workspace layout (bytes):
    //   Ah: 0                (16384*1024*2 = 33,554,432)
    //   Al: 33,554,432
    //   Bh: 67,108,864       (1000*1024*2 = 2,048,000)
    //   Bl: 69,156,864
    //   anorm: 71,204,864    (16384*4)
    //   bnorm: 71,270,400    (1000*4)    -> total ~71.3 MB
    char* ws = (char*)d_ws;
    unsigned short* Ah = (unsigned short*)(ws);
    unsigned short* Al = (unsigned short*)(ws + 33554432u);
    unsigned short* Bh = (unsigned short*)(ws + 67108864u);
    unsigned short* Bl = (unsigned short*)(ws + 69156864u);
    float* anorm = (float*)(ws + 71204864u);
    float* bnorm = (float*)(ws + 71270400u);

    // One wave per row; 8 rows per 256-thread block.
    split_norm_kernel<<<DIM_N / 8, 256, 0, stream>>>(emb, Ah, Al, anorm, DIM_N);
    split_norm_kernel<<<(DIM_C + 7) / 8, 256, 0, stream>>>(cen, Bh, Bl, bnorm, DIM_C);

    dim3 grid(DIM_N / 256, 1024 / 64);          // (64, 16)
    dist_wmma_kernel<<<grid, 128, 0, stream>>>(Ah, Al, Bh, Bl, anorm, bnorm, out);
}
